// MultiHeadAttention_56083682951179
// MI455X (gfx1250) — matrix operand order repaired and checked by reading, not compiled, not run on hardware
//
#include <hip/hip_runtime.h>
#include <hip/hip_bf16.h>

// ---------------------------------------------------------------------------
// MHA for MI455X (gfx1250): bf16 WMMA pipeline.
//   B=2, N=2048, D=1024, H=16, Dh=64, scale=0.125
// Kernels:
//   1. convert_wt     : Wq/Wk/Wv fp32 [H,D,Dh] -> bf16 transposed [H,Dh,D]
//   2. convert_wo     : Wo fp32 [D,D] -> bf16 [D,D]
//   3. proj_kernel    : X[b] @ W[h] -> qh/kh (row-major) and vt (transposed)
//   4. attn_kernel    : flash attention with online softmax, WMMA QK^T + PV
//   5. outproj_kernel : concat @ Wo^T -> fp32 out
//
// Round-3 changes: explicit 2-stage software pipelines (manual unroll x2 with
// two live register sets) so WMMAs consume fragments loaded a full stage
// earlier -> partial s_wait_loadcnt instead of 0x0 before each WMMA.
// Prefetch locality 3 (WGP scope) instead of SYS.
// ---------------------------------------------------------------------------

typedef __bf16 bf16;
typedef __attribute__((ext_vector_type(16))) __bf16 v16bf;
typedef __attribute__((ext_vector_type(8)))  __bf16 v8bf;
typedef __attribute__((ext_vector_type(8)))  float  v8f;
typedef __attribute__((ext_vector_type(4)))  float  v4f;

#define B_    2
#define N_    2048
#define D_    1024
#define H_    16
#define DH_   64
#define SCALE_ 0.125f

static __device__ __forceinline__ v16bf cat16(v8bf a, v8bf b) {
  return __builtin_shufflevector(a, b, 0,1,2,3,4,5,6,7,8,9,10,11,12,13,14,15);
}

static __device__ __forceinline__ v8f wmma_bf16(v16bf a, v16bf b, v8f c) {
  // (neg_a, A, neg_b, B, c_mod, C, reuse_a, reuse_b)
  return __builtin_amdgcn_wmma_f32_16x16x32_bf16(false, a, false, b, (short)0, c,
                                                 false, false);
}

// A-fragment (16x32, 16-bit): lane row = lane%16; K per lane (half = lane/16):
//   j in [0,15] -> K = half*16 + j       (contiguous run of 16, matches B)
static __device__ __forceinline__ v16bf load_a_bf16(const bf16* row_k0, int half) {
  v8bf lo = *(const v8bf*)(row_k0 + half * 16);
  v8bf hi = *(const v8bf*)(row_k0 + half * 16 + 8);
  return cat16(lo, hi);
}

// B-fragment (32x16, 16-bit): lane col = lane%16; K = half*16 + j
// caller pre-offsets pointer by half*16; 16 contiguous bf16.
static __device__ __forceinline__ v16bf load_b_bf16(const bf16* p) {
  v8bf lo = *(const v8bf*)(p);
  v8bf hi = *(const v8bf*)(p + 8);
  return cat16(lo, hi);
}

// --------------------------- weight prep -----------------------------------

__global__ __launch_bounds__(256) void convert_wt_kernel(
    const float* __restrict__ W, bf16* __restrict__ Wt) {
  // W: [H, D, Dh] fp32  ->  Wt: [H, Dh, D] bf16   (H*D*Dh = 1<<20 elements)
  int idx = blockIdx.x * blockDim.x + threadIdx.x;
  int h   = idx >> 16;            // D*Dh = 65536
  int rem = idx & 0xFFFF;
  int d   = rem >> 6;
  int e   = rem & 63;
  Wt[((size_t)h << 16) + (size_t)e * D_ + d] = (bf16)W[idx];
}

__global__ __launch_bounds__(256) void convert_wo_kernel(
    const float* __restrict__ W, bf16* __restrict__ Wt) {
  int idx = blockIdx.x * blockDim.x + threadIdx.x;
  Wt[idx] = (bf16)W[idx];
}

// --------------------------- projections -----------------------------------

__global__ __launch_bounds__(256) void proj_kernel(
    const float* __restrict__ X,   // [B, N, D] fp32
    const bf16*  __restrict__ Wt,  // [H, Dh, D] bf16 (transposed)
    bf16* __restrict__ Y,          // [B*H, N, Dh] if !transposed else [B*H, Dh, N]
    int transposed) {
  const int lane = threadIdx.x & 31;
  const int wv   = threadIdx.x >> 5;
  const int half = lane >> 4;
  const int l16  = lane & 15;
  const int bh   = blockIdx.y;
  const int b    = bh / H_;
  const int h    = bh % H_;
  const int row0 = blockIdx.x * 128 + wv * 16;

  const float* xr = X + ((size_t)b * N_ + row0 + l16) * D_;
  const bf16*  Wh = Wt + (size_t)h * DH_ * D_;

  v8f acc[4];
  #pragma unroll
  for (int f = 0; f < 4; ++f)
    #pragma unroll
    for (int r = 0; r < 8; ++r) acc[f][r] = 0.f;

  // A fragment from fp32 activations: two runs of 8 floats, converted to bf16
  auto loadA = [&](int k0) -> v16bf {
    const float* p1 = xr + k0 + half * 16;
    const float* p2 = p1 + 8;
    v4f a0 = *(const v4f*)(p1);
    v4f a1 = *(const v4f*)(p1 + 4);
    v4f a2 = *(const v4f*)(p2);
    v4f a3 = *(const v4f*)(p2 + 4);
    v16bf A;
    #pragma unroll
    for (int i = 0; i < 4; ++i) {
      A[i]      = (__bf16)a0[i];
      A[4 + i]  = (__bf16)a1[i];
      A[8 + i]  = (__bf16)a2[i];
      A[12 + i] = (__bf16)a3[i];
    }
    return A;
  };
  auto loadB4 = [&](int k0, v16bf* Bv) {
    #pragma unroll
    for (int f = 0; f < 4; ++f)
      Bv[f] = load_b_bf16(Wh + (size_t)(f * 16 + l16) * D_ + k0 + half * 16);
  };

  // ---- 2-stage software pipeline over K (step 32, unrolled x2) ----
  v16bf A0, A1, B0[4], B1[4];
  A0 = loadA(0);
  loadB4(0, B0);
  for (int k0 = 0; k0 < D_; k0 += 64) {
    A1 = loadA(k0 + 32);
    loadB4(k0 + 32, B1);
    __builtin_prefetch(xr + k0 + 64 + half * 8, 0, 3);
    #pragma unroll
    for (int f = 0; f < 4; ++f) acc[f] = wmma_bf16(A0, B0[f], acc[f]);
    if (k0 + 64 < D_) {
      A0 = loadA(k0 + 64);
      loadB4(k0 + 64, B0);
    }
    #pragma unroll
    for (int f = 0; f < 4; ++f) acc[f] = wmma_bf16(A1, B1[f], acc[f]);
  }

  #pragma unroll
  for (int f = 0; f < 4; ++f)
    #pragma unroll
    for (int r = 0; r < 8; ++r) {
      int n = row0 + r + 8 * half;
      int e = f * 16 + l16;
      bf16 v = (bf16)acc[f][r];
      if (!transposed) Y[((size_t)bh * N_ + n) * DH_ + e] = v;
      else             Y[((size_t)bh * DH_ + e) * N_ + n] = v;
    }
}

// --------------------------- flash attention -------------------------------

__global__ __launch_bounds__(256) void attn_kernel(
    const bf16* __restrict__ qh,   // [B*H, N, Dh]
    const bf16* __restrict__ kh,   // [B*H, N, Dh]
    const bf16* __restrict__ vt,   // [B*H, Dh, N]  (V transposed)
    bf16* __restrict__ concat) {   // [B, N, D]  head h -> cols [h*Dh, h*Dh+64)
  // wave-private P staging: 16 rows x 32 keys, padded to 48 bf16/row (96 B,
  // multiple of 16 B so ds_read_b128 stays aligned)
  __shared__ __bf16 pbuf[8 * 16 * 48];

  const int lane = threadIdx.x & 31;
  const int wv   = threadIdx.x >> 5;
  const int half = lane >> 4;
  const int l16  = lane & 15;
  const int bh   = blockIdx.y;
  const int b    = bh / H_;
  const int h    = bh % H_;
  const int q0   = blockIdx.x * 128 + wv * 16;

  const bf16* qb = qh + (size_t)bh * N_ * DH_;
  const bf16* kb = kh + (size_t)bh * N_ * DH_;
  const bf16* vb = vt + (size_t)bh * DH_ * N_;
  __bf16* pw = pbuf + wv * (16 * 48);

  // preload Q fragments for the whole row-tile (K = 0..31 and 32..63)
  const bf16* qrow = qb + (size_t)(q0 + l16) * DH_;
  v16bf Aq[2];
  #pragma unroll
  for (int kk = 0; kk < 2; ++kk) Aq[kk] = load_a_bf16(qrow + kk * 32, half);

  float mrow[8], lrow[8];
  #pragma unroll
  for (int r = 0; r < 8; ++r) { mrow[r] = -__builtin_inff(); lrow[r] = 0.f; }
  v8f acc[4];
  #pragma unroll
  for (int f = 0; f < 4; ++f)
    #pragma unroll
    for (int r = 0; r < 8; ++r) acc[f][r] = 0.f;

  auto loadK = [&](int m0, v16bf (&Kf)[2][2]) {
    #pragma unroll
    for (int t = 0; t < 2; ++t)
      #pragma unroll
      for (int kk = 0; kk < 2; ++kk)
        Kf[t][kk] = load_b_bf16(kb + (size_t)(m0 + t * 16 + l16) * DH_ +
                                kk * 32 + half * 16);
  };
  auto loadV = [&](int m0, v16bf* Vf) {
    #pragma unroll
    for (int f = 0; f < 4; ++f)
      Vf[f] = load_b_bf16(vb + (size_t)(f * 16 + l16) * N_ + m0 + half * 16);
  };

  // one 32-key block: scores (4 WMMA), online softmax, LDS reshape, PV (4 WMMA)
  auto process = [&](int m0, v16bf (&Kf)[2][2], v16bf* Vf) {
    v8f S[2];
    #pragma unroll
    for (int t = 0; t < 2; ++t) {
      #pragma unroll
      for (int r = 0; r < 8; ++r) S[t][r] = 0.f;
      #pragma unroll
      for (int kk = 0; kk < 2; ++kk) S[t] = wmma_bf16(Aq[kk], Kf[t][kk], S[t]);
    }
    // C/D layout: VGPR r holds row (r + 8*half), cols striped on lane%16,
    // so xor-shuffles with masks 1/2/4/8 reduce within each 16-lane half.
    float corr[8];
    #pragma unroll
    for (int r = 0; r < 8; ++r) {
      float s0 = S[0][r] * SCALE_;
      float s1 = S[1][r] * SCALE_;
      float mx = fmaxf(s0, s1);
      #pragma unroll
      for (int d = 1; d <= 8; d <<= 1) mx = fmaxf(mx, __shfl_xor(mx, d, 32));
      float mn = fmaxf(mrow[r], mx);
      corr[r] = __expf(mrow[r] - mn);
      float p0 = __expf(s0 - mn);
      float p1 = __expf(s1 - mn);
      float sum = p0 + p1;
      #pragma unroll
      for (int d = 1; d <= 8; d <<= 1) sum += __shfl_xor(sum, d, 32);
      lrow[r] = lrow[r] * corr[r] + sum;
      mrow[r] = mn;
      // stage P tile (D-layout) into LDS for A-layout reload
      pw[(r + 8 * half) * 48 + l16]      = (__bf16)p0;
      pw[(r + 8 * half) * 48 + 16 + l16] = (__bf16)p1;
    }
    #pragma unroll
    for (int f = 0; f < 4; ++f)
      #pragma unroll
      for (int r = 0; r < 8; ++r) acc[f][r] *= corr[r];
    // LDS is wave-private: wave-local ds write->read hazard only
    asm volatile("s_wait_dscnt 0x0" ::: "memory");
    v16bf Ap = load_a_bf16(pw + l16 * 48, half);
    #pragma unroll
    for (int f = 0; f < 4; ++f) acc[f] = wmma_bf16(Ap, Vf[f], acc[f]);
  };

  // ---- key-block loop, K double-buffered across blocks (unroll x2) ----
  v16bf Ka[2][2], Kc[2][2];
  loadK(0, Ka);
  for (int m0 = 0; m0 < N_; m0 += 64) {
    v16bf Vf[4];
    loadV(m0, Vf);          // V latency hides behind softmax VALU work
    loadK(m0 + 32, Kc);     // next block's K in flight through this block
    process(m0, Ka, Vf);
    loadV(m0 + 32, Vf);
    if (m0 + 64 < N_) {
      loadK(m0 + 64, Ka);
      __builtin_prefetch(kb + (size_t)(m0 + 96 + l16) * DH_, 0, 3);
      __builtin_prefetch(vb + (size_t)l16 * N_ + m0 + 96 + half * 16, 0, 3);
    }
    process(m0 + 32, Kc, Vf);
  }

  // normalize and write to concat buffer (bf16) at this head's column slice
  float inv[8];
  #pragma unroll
  for (int r = 0; r < 8; ++r) inv[r] = 1.f / lrow[r];
  bf16* cb = concat + (size_t)b * N_ * D_;
  #pragma unroll
  for (int f = 0; f < 4; ++f)
    #pragma unroll
    for (int r = 0; r < 8; ++r) {
      int n = q0 + r + 8 * half;
      int c = h * DH_ + f * 16 + l16;
      cb[(size_t)n * D_ + c] = (bf16)(acc[f][r] * inv[r]);
    }
}

// --------------------------- output projection -----------------------------

__global__ __launch_bounds__(256) void outproj_kernel(
    const bf16* __restrict__ concat,  // [B*N, D] bf16
    const bf16* __restrict__ Wo,      // [D, D] bf16 (row o has contiguous d)
    float* __restrict__ out) {        // [B*N, D] fp32
  const int lane = threadIdx.x & 31;
  const int wv   = threadIdx.x >> 5;
  const int half = lane >> 4;
  const int l16  = lane & 15;
  const int wg   = blockIdx.x * 8 + wv;     // global wave-tile id
  const int row0 = (wg >> 4) * 16;          // 256 row strips of 16
  const int c0   = (wg & 15) * 64;          // 16 col strips of 64

  v8f acc[4];
  #pragma unroll
  for (int f = 0; f < 4; ++f)
    #pragma unroll
    for (int r = 0; r < 8; ++r) acc[f][r] = 0.f;

  const bf16* ar = concat + (size_t)(row0 + l16) * D_;
  auto loadB4 = [&](int k0, v16bf* Bv) {
    // out[n][o] = sum_d concat[n][d] * Wo[o][d]  -> B[k=d][n=o] = Wo[o][d]
    #pragma unroll
    for (int f = 0; f < 4; ++f)
      Bv[f] = load_b_bf16(Wo + (size_t)(c0 + f * 16 + l16) * D_ + k0 + half * 16);
  };

  // ---- 2-stage software pipeline over K (step 32, unrolled x2) ----
  v16bf A0, A1, B0[4], B1[4];
  A0 = load_a_bf16(ar, half);
  loadB4(0, B0);
  for (int k0 = 0; k0 < D_; k0 += 64) {
    A1 = load_a_bf16(ar + k0 + 32, half);
    loadB4(k0 + 32, B1);
    __builtin_prefetch(ar + k0 + 64 + half * 8, 0, 3);
    #pragma unroll
    for (int f = 0; f < 4; ++f) acc[f] = wmma_bf16(A0, B0[f], acc[f]);
    if (k0 + 64 < D_) {
      A0 = load_a_bf16(ar + k0 + 64, half);
      loadB4(k0 + 64, B0);
    }
    #pragma unroll
    for (int f = 0; f < 4; ++f) acc[f] = wmma_bf16(A1, B1[f], acc[f]);
  }

  #pragma unroll
  for (int f = 0; f < 4; ++f)
    #pragma unroll
    for (int r = 0; r < 8; ++r)
      out[(size_t)(row0 + r + 8 * half) * D_ + c0 + f * 16 + l16] = acc[f][r];
}

// --------------------------- launcher --------------------------------------

extern "C" void kernel_launch(void* const* d_in, const int* in_sizes, int n_in,
                              void* d_out, int out_size, void* d_ws, size_t ws_size,
                              hipStream_t stream) {
  const float* q  = (const float*)d_in[0];
  const float* k  = (const float*)d_in[1];
  const float* v  = (const float*)d_in[2];
  const float* Wq = (const float*)d_in[3];
  const float* Wk = (const float*)d_in[4];
  const float* Wv = (const float*)d_in[5];
  const float* Wo = (const float*)d_in[6];
  float* out = (float*)d_out;

  // workspace carve-up (all offsets multiples of 2 MB -> 16B alignment holds)
  const size_t WSZ = (size_t)H_ * DH_ * D_;       // 1<<20 elems per weight
  const size_t PSZ = (size_t)B_ * H_ * N_ * DH_;  // 4<<20 elems per tensor
  char* ws = (char*)d_ws;
  bf16* wq_t = (bf16*)ws;  ws += WSZ * 2;
  bf16* wk_t = (bf16*)ws;  ws += WSZ * 2;
  bf16* wv_t = (bf16*)ws;  ws += WSZ * 2;
  bf16* wo_b = (bf16*)ws;  ws += (size_t)D_ * D_ * 2;
  bf16* qh   = (bf16*)ws;  ws += PSZ * 2;
  bf16* kh   = (bf16*)ws;  ws += PSZ * 2;
  bf16* vt   = (bf16*)ws;  ws += PSZ * 2;
  bf16* cc   = (bf16*)ws;  // concat [B, N, D]

  dim3 blk(256);
  convert_wt_kernel<<<(unsigned)(WSZ / 256), blk, 0, stream>>>(Wq, wq_t);
  convert_wt_kernel<<<(unsigned)(WSZ / 256), blk, 0, stream>>>(Wk, wk_t);
  convert_wt_kernel<<<(unsigned)(WSZ / 256), blk, 0, stream>>>(Wv, wv_t);
  convert_wo_kernel<<<(unsigned)((size_t)D_ * D_ / 256), blk, 0, stream>>>(Wo, wo_b);

  dim3 pgrid(N_ / 128, B_ * H_);
  proj_kernel<<<pgrid, blk, 0, stream>>>(q, wq_t, qh, 0);
  proj_kernel<<<pgrid, blk, 0, stream>>>(k, wk_t, kh, 0);
  proj_kernel<<<pgrid, blk, 0, stream>>>(v, wv_t, vt, 1);

  attn_kernel<<<dim3(N_ / 128, B_ * H_), blk, 0, stream>>>(qh, kh, vt, cc);

  outproj_kernel<<<(B_ * N_ / 16) * (D_ / 64) / 8, blk, 0, stream>>>(cc, wo_b, out);
}
